// SpikingNetwork_78778290143907
// MI455X (gfx1250) — matrix operand order repaired and checked by reading, not compiled, not run on hardware
//
#include <hip/hip_runtime.h>
#include <hip/hip_bf16.h>
#include <math.h>

typedef __attribute__((ext_vector_type(16))) _Float16 v16h;
typedef __attribute__((ext_vector_type(8)))  float    v8f;

namespace {
constexpr int L0 = 4096, L1 = 8192, L2 = 8192, L3 = 10;
constexpr int C0 = 4096, C1 = 4096, C2 = 5;   // columns of weights0/1/2
constexpr int PAD3 = 16;                       // output layer padded 10 -> 16
constexpr int MAX_STEPS = 50;
constexpr int ROWG = 64, COLCH = 4;            // 256 blocks for streaming kernels
}

// ---------------- init / preprocessing ----------------

__global__ void k_zero(float4* __restrict__ p, size_t n4) {
  size_t i  = (size_t)blockIdx.x * blockDim.x + threadIdx.x;
  size_t st = (size_t)gridDim.x * blockDim.x;
  float4 z = {0.f, 0.f, 0.f, 0.f};
  for (; i < n4; i += st) p[i] = z;
}

__global__ void k_init_state(float* p1, float* p2, float* p3,
                             int* cnt1, int* cnt2, int* done,
                             int* fired, int* times,
                             int* list1, int* list2) {
  int i = blockIdx.x * blockDim.x + threadIdx.x;
  if (i < L1) { p1[i] = 0.f; list1[i] = 0; }   // lists pre-filled with valid
  if (i < L2) { p2[i] = 0.f; list2[i] = 0; }   // row indices (see k_out_wmma)
  if (i < PAD3) { p3[i] = 0.f; fired[i] = 0; times[i] = -1; }
  if (i == 0) { *cnt1 = 0; *cnt2 = 0; *done = 0; }
}

// W_eff[row][tgt] += w[row][j]  (targets collide within a row -> atomics)
__global__ void k_build_scatter(const float* __restrict__ w,
                                const int* __restrict__ tgt,
                                float* __restrict__ W,
                                size_t nElems, int colsW, int stride) {
  size_t i  = (size_t)blockIdx.x * blockDim.x + threadIdx.x;
  size_t st = (size_t)gridDim.x * blockDim.x;
  for (; i < nElems; i += st) {
    int row = (int)(i / (unsigned)colsW);
    atomicAdd(&W[(size_t)row * stride + tgt[i]], w[i]);
  }
}

__global__ void k_convert_h(const float* __restrict__ src,
                            _Float16* __restrict__ dst, int n) {
  int i = blockIdx.x * blockDim.x + threadIdx.x;
  if (i < n) dst[i] = (_Float16)src[i];
}

// ---------------- per-step streaming accumulation ----------------

// t==0 only: p1 += (2*input[r]) * W0_eff[r][:], dense over all 4096 rows.
__global__ void k_accum_dense0(const float* __restrict__ inp,
                               const float* __restrict__ W,
                               float* __restrict__ pOut,
                               const int* __restrict__ maxT) {
  if (*maxT < 1) return;
  const int colChunk = blockIdx.x & (COLCH - 1);
  const int rowGroup = blockIdx.x >> 2;
  const int colBase  = colChunk * 2048 + threadIdx.x * 8;
  float a0 = 0.f, a1 = 0.f, a2 = 0.f, a3 = 0.f, a4 = 0.f, a5 = 0.f, a6 = 0.f, a7 = 0.f;
  for (int r = rowGroup; r < L0; r += ROWG) {
    float s = 2.0f * inp[r];
    const float4* row = (const float4*)(W + (size_t)r * L1 + colBase);
    if ((threadIdx.x & 31) == 0 && r + ROWG < L0)
      __builtin_prefetch(W + (size_t)(r + ROWG) * L1 + colBase, 0, 1);
    float4 w0 = row[0], w1 = row[1];
    a0 += s * w0.x; a1 += s * w0.y; a2 += s * w0.z; a3 += s * w0.w;
    a4 += s * w1.x; a5 += s * w1.y; a6 += s * w1.z; a7 += s * w1.w;
  }
  atomicAdd(&pOut[colBase + 0], a0); atomicAdd(&pOut[colBase + 1], a1);
  atomicAdd(&pOut[colBase + 2], a2); atomicAdd(&pOut[colBase + 3], a3);
  atomicAdd(&pOut[colBase + 4], a4); atomicAdd(&pOut[colBase + 5], a5);
  atomicAdd(&pOut[colBase + 6], a6); atomicAdd(&pOut[colBase + 7], a7);
}

// Binary-spike layers: p += sum of W_eff rows listed in `list` (row-sum).
__global__ void k_accum_rows(const float* __restrict__ W, int stride,
                             const int* __restrict__ list,
                             const int* __restrict__ cnt,
                             float* __restrict__ pOut,
                             const int* __restrict__ done,
                             const int* __restrict__ maxT, int t) {
  if (*done || t >= *maxT) return;
  const int count = *cnt;
  const int colChunk = blockIdx.x & (COLCH - 1);
  const int rowGroup = blockIdx.x >> 2;
  if (rowGroup >= count) return;
  const int colBase = colChunk * 2048 + threadIdx.x * 8;
  float a0 = 0.f, a1 = 0.f, a2 = 0.f, a3 = 0.f, a4 = 0.f, a5 = 0.f, a6 = 0.f, a7 = 0.f;
  for (int r = rowGroup; r < count; r += ROWG) {
    const float4* row = (const float4*)(W + (size_t)list[r] * stride + colBase);
    if ((threadIdx.x & 31) == 0 && r + ROWG < count)
      __builtin_prefetch(W + (size_t)list[r + ROWG] * stride + colBase, 0, 1);
    float4 w0 = row[0], w1 = row[1];
    a0 += w0.x; a1 += w0.y; a2 += w0.z; a3 += w0.w;
    a4 += w1.x; a5 += w1.y; a6 += w1.z; a7 += w1.w;
  }
  atomicAdd(&pOut[colBase + 0], a0); atomicAdd(&pOut[colBase + 1], a1);
  atomicAdd(&pOut[colBase + 2], a2); atomicAdd(&pOut[colBase + 3], a3);
  atomicAdd(&pOut[colBase + 4], a4); atomicAdd(&pOut[colBase + 5], a5);
  atomicAdd(&pOut[colBase + 6], a6); atomicAdd(&pOut[colBase + 7], a7);
}

// Output layer 8192 -> 10 via v_wmma_f32_16x16x32_f16 (single wave).
// A = binary spike validity (exact in f16), B = gathered W2_eff f16 tiles.
// All loads are unconditional: list[] always holds valid row indices
// (zero-initialized; stale entries are old valid indices), and padded K
// slots are zeroed through the A matrix instead of load predication.
__global__ void k_out_wmma(const _Float16* __restrict__ W2h,
                           const int* __restrict__ list,
                           const int* __restrict__ cnt,
                           float* __restrict__ p3,
                           const int* __restrict__ done,
                           const int* __restrict__ maxT, int t) {
  if (*done || t >= *maxT) return;
  const int count = *cnt;
  if (count == 0) return;
  const int lane = threadIdx.x;      // 32 threads, EXEC all ones at WMMA
  const int half = lane >> 4;
  const int n    = lane & 15;
  v8f c = {};

  v16h aFull;
#pragma unroll
  for (int s = 0; s < 16; ++s) aFull[s] = (_Float16)1.0f;

  const int nChunks = (count + 31) >> 5;
  for (int ch = 0; ch < nChunks; ++ch) {
    const int kb = ch << 5;
    // one coalesced list read per lane; distribute with ds_bpermute
    int pos   = kb + lane;
    int myRow = list[pos < count ? pos : (count - 1)];

    v16h a;
    if (kb + 32 <= count) {
      a = aFull;                     // steady state: all 32 spikes valid
    } else {
#pragma unroll
      for (int s = 0; s < 16; ++s) {
        // A K order matches B: lanes 0-15 hold K in {0..15},
        // lanes 16-31 hold K in {16..31}
        int K = s + (half ? 16 : 0);
        a[s] = (kb + K < count) ? (_Float16)1.0f : (_Float16)0.0f;
      }
    }

    v16h b;
#pragma unroll
    for (int v = 0; v < 8; ++v) {
      // B 32x16: VGPR v, lanes 0-15 -> rows 2v,2v+1 ; lanes 16-31 -> +16
      int src = (half ? 16 : 0) + 2 * v;
      int r0 = __shfl(myRow, src, 32);
      int r1 = __shfl(myRow, src + 1, 32);
      b[2 * v]     = W2h[(size_t)r0 * PAD3 + n];
      b[2 * v + 1] = W2h[(size_t)r1 * PAD3 + n];
    }
    c = __builtin_amdgcn_wmma_f32_16x16x32_f16(false, a, false, b, (short)0, c,
                                               false, false);
  }
  // D row M=0 lives in VGPR0 of lanes 0-15 (N = lane)
  if (lane < L3) atomicAdd(&p3[lane], c[0]);
}

// ---------------- per-step decay / threshold / compaction ----------------

__global__ void k_update_hidden(float* __restrict__ p, int* __restrict__ list,
                                int* __restrict__ cnt,
                                const int* __restrict__ done,
                                const int* __restrict__ maxT,
                                int t, float decay, int n) {
  if (*done || t >= *maxT) return;
  __shared__ int lcnt;
  if (threadIdx.x == 0) lcnt = 0;
  __syncthreads();
  for (int i = threadIdx.x; i < n; i += blockDim.x) {
    float v = p[i] * decay;
    if (v >= 1.0f) {                 // spike -> reset, record index
      p[i] = 0.0f;
      int idx = atomicAdd(&lcnt, 1); // LDS atomic (ds_add_rtn)
      list[idx] = i;
    } else {
      p[i] = v;
    }
  }
  __syncthreads();
  if (threadIdx.x == 0) *cnt = lcnt;
}

__global__ void k_update_out(float* __restrict__ p3, int* __restrict__ fired,
                             int* __restrict__ times, int* __restrict__ done,
                             const int* __restrict__ maxT, int t, float decay) {
  if (*done || t >= *maxT) return;
  __shared__ int nf;
  int lane = threadIdx.x;            // 32 threads
  if (lane == 0) nf = 0;
  __syncthreads();
  if (lane < L3) {
    float v = p3[lane] * decay;
    p3[lane] = v;                    // output layer is not reset
    int above = (v >= 1.0f) ? 1 : 0;
    int f = fired[lane];
    if (above && !f) times[lane] = t;
    f |= above;
    fired[lane] = f;
    if (f) atomicAdd(&nf, 1);
  }
  __syncthreads();
  if (lane == 0 && nf == L3) *done = 1;   // freeze everything from next step
}

__global__ void k_writeout(const int* __restrict__ times,
                           const float* __restrict__ p3,
                           float* __restrict__ out) {
  int lane = threadIdx.x;
  if (lane < L3) {
    out[lane]      = (float)times[lane]; // output_spike_times
    out[L3 + lane] = p3[lane];           // final output potentials
  }
}

// ---------------- host orchestration (graph-capture safe) ----------------

extern "C" void kernel_launch(void* const* d_in, const int* in_sizes, int n_in,
                              void* d_out, int out_size, void* d_ws, size_t ws_size,
                              hipStream_t stream) {
  (void)in_sizes; (void)n_in; (void)out_size; (void)ws_size;
  const float* inp = (const float*)d_in[0];
  const float* w0  = (const float*)d_in[1];
  const float* w1  = (const float*)d_in[2];
  const float* w2  = (const float*)d_in[3];
  const int*   t0  = (const int*)d_in[4];
  const int*   t1  = (const int*)d_in[5];
  const int*   t2  = (const int*)d_in[6];
  const int*   dMaxT = (const int*)d_in[7];
  float* out = (float*)d_out;

  char* base = (char*)d_ws;
  size_t off = 0;
  auto alloc = [&](size_t b) -> char* {
    char* p = base + off;
    off = (off + b + 255) & ~(size_t)255;
    return p;
  };
  float*    W0   = (float*)alloc((size_t)L0 * L1 * 4);     // 128 MB
  float*    W1   = (float*)alloc((size_t)L1 * L2 * 4);     // 256 MB
  float*    W2f  = (float*)alloc((size_t)L2 * PAD3 * 4);
  _Float16* W2h  = (_Float16*)alloc((size_t)L2 * PAD3 * 2);
  float*    p1   = (float*)alloc((size_t)L1 * 4);
  float*    p2   = (float*)alloc((size_t)L2 * 4);
  float*    p3   = (float*)alloc((size_t)PAD3 * 4);
  int*      list1 = (int*)alloc((size_t)L1 * 4);
  int*      list2 = (int*)alloc((size_t)L2 * 4);
  int*      cnt1 = (int*)alloc(4);
  int*      cnt2 = (int*)alloc(4);
  int*      done = (int*)alloc(4);
  int*      fired = (int*)alloc((size_t)PAD3 * 4);
  int*      times = (int*)alloc((size_t)PAD3 * 4);

  const float decay = expf(-1.0f / 20.0f);

  // Zero the scatter targets (rebuilt every call: determinism requirement).
  k_zero<<<4096, 256, 0, stream>>>((float4*)W0,  (size_t)L0 * L1 / 4);
  k_zero<<<4096, 256, 0, stream>>>((float4*)W1,  (size_t)L1 * L2 / 4);
  k_zero<<<64,   256, 0, stream>>>((float4*)W2f, (size_t)L2 * PAD3 / 4);
  k_init_state<<<32, 256, 0, stream>>>(p1, p2, p3, cnt1, cnt2, done, fired,
                                       times, list1, list2);

  // Build dense effective matrices (amortized over all timesteps).
  k_build_scatter<<<4096, 256, 0, stream>>>(w0, t0, W0, (size_t)L0 * C0, C0, L1);
  k_build_scatter<<<4096, 256, 0, stream>>>(w1, t1, W1, (size_t)L1 * C1, C1, L2);
  k_build_scatter<<<256,  256, 0, stream>>>(w2, t2, W2f, (size_t)L2 * C2, C2, PAD3);
  k_convert_h<<<(L2 * PAD3 + 255) / 256, 256, 0, stream>>>(W2f, W2h, L2 * PAD3);

  for (int t = 0; t < MAX_STEPS; ++t) {
    if (t == 0)  // input spikes only active at t=0
      k_accum_dense0<<<ROWG * COLCH, 256, 0, stream>>>(inp, W0, p1, dMaxT);
    k_accum_rows<<<ROWG * COLCH, 256, 0, stream>>>(W1, L2, list1, cnt1, p2,
                                                   done, dMaxT, t);
    k_out_wmma<<<1, 32, 0, stream>>>(W2h, list2, cnt2, p3, done, dMaxT, t);
    k_update_hidden<<<1, 1024, 0, stream>>>(p1, list1, cnt1, done, dMaxT, t,
                                            decay, L1);
    k_update_hidden<<<1, 1024, 0, stream>>>(p2, list2, cnt2, done, dMaxT, t,
                                            decay, L2);
    k_update_out<<<1, 32, 0, stream>>>(p3, fired, times, done, dMaxT, t, decay);
  }
  k_writeout<<<1, 32, 0, stream>>>(times, p3, out);
}
